// GMM_9242769620998
// MI455X (gfx1250) — matrix operand order repaired and checked by reading, not compiled, not run on hardware
//
#include <hip/hip_runtime.h>
#include <hip/hip_bf16.h>

typedef __attribute__((ext_vector_type(16))) __bf16        v16bf;
typedef __attribute__((ext_vector_type(8)))  float         v8f;
typedef __attribute__((ext_vector_type(8)))  unsigned int  u32x8;

__device__ __forceinline__ unsigned short f32_to_bf16(float f) {
    unsigned int u = __builtin_bit_cast(unsigned int, f);
    unsigned int r = u + 0x7FFFu + ((u >> 16) & 1u);   // round-to-nearest-even
    return (unsigned short)(r >> 16);
}

// Assemble a bf16 16-wide fragment from two 16B LDS reads (ds_load_b128).
__device__ __forceinline__ v16bf frag_from_lds(const unsigned int* rowp, int p0, int p1) {
    uint4 lo = *(const uint4*)(rowp + p0);
    uint4 hi = *(const uint4*)(rowp + p1);
    u32x8 u;
    u[0] = lo.x; u[1] = lo.y; u[2] = lo.z; u[3] = lo.w;
    u[4] = hi.x; u[5] = hi.y; u[6] = hi.z; u[7] = hi.w;
    return __builtin_bit_cast(v16bf, u);
}

// Async global->LDS 16B copy (CDNA5 GLOBAL_LOAD_ASYNC_TO_LDS_B128, ASYNCcnt).
// ldsAddr: wave-relative LDS byte address (low 32 bits of generic shared ptr).
__device__ __forceinline__ void async_copy_b128(unsigned ldsAddr, const void* gaddr) {
    asm volatile("global_load_async_to_lds_b128 %0, %1, off"
                 :: "v"(ldsAddr), "v"((unsigned long long)gaddr)
                 : "memory");
}
__device__ __forceinline__ void wait_async0() {
    asm volatile("s_wait_asynccnt 0x0" ::: "memory");
}

// ---------------------------------------------------------------------------
// Generic bf16 WMMA GEMM:  out = epilogue(A[M,K] * W^T[N,K] + bias) (+resid)
// Block: 256 thr = 8 waves; block tile 32x256; wave tile 16x64; K step 64.
// LDS rows padded to 36 dwords: 16B-aligned b128 access, conflict-free.
// Staging uses async global->LDS copies (no VGPR round-trip).
// epi: 0=none 1=silu 2=elu+1 3=gelu(exact) 4=kv-split(elu+1 | identity)
// ---------------------------------------------------------------------------
__global__ __launch_bounds__(256)
void k_gemm_bf16(const unsigned int* __restrict__ A32,   // [M, K/2] packed bf16 pairs
                 const unsigned int* __restrict__ W32,   // [N, K/2] packed bf16 pairs
                 const float*  __restrict__ bias,
                 float*        __restrict__ outF,
                 float*        __restrict__ outF2,       // v-half for kv split
                 unsigned short* __restrict__ outB,      // bf16 output (fc1)
                 const float*  __restrict__ resid,
                 int M, int N, int K, int epi)
{
    __shared__ __align__(16) unsigned int la[32][36];    // 32 rows x 32 dw (+4 pad)
    __shared__ __align__(16) unsigned int lb[256][36];   // 256 rows x 32 dw (+4 pad)

    const int tid  = threadIdx.x;
    const int lane = tid & 31;
    const int wave = tid >> 5;
    const int wm   = wave >> 2;        // 0..1
    const int wn   = wave & 3;         // 0..3
    const int half = lane >> 4;        // 0..1
    const int l16  = lane & 15;        // 0..15
    const int m0   = blockIdx.x * 32;
    const int n0   = blockIdx.y * 256;
    const int K2   = K >> 1;

    // staging coordinates (one uint4 = 4 dwords per chunk)
    const int arow = tid >> 3;             // 0..31
    const int acc4 = (tid & 7) * 4;        // 0,4,..,28

    const unsigned laAddr = (unsigned)(size_t)&la[arow][acc4];
    unsigned lbAddr[8];
    int lbRow[8], lbCol[8];
#pragma unroll
    for (int j = 0; j < 8; ++j) {
        int chunk = tid + j * 256;
        lbRow[j] = chunk >> 3;
        lbCol[j] = (chunk & 7) * 4;
        lbAddr[j] = (unsigned)(size_t)&lb[lbRow[j]][lbCol[j]];
    }

    v8f acc[4];
#pragma unroll
    for (int nt = 0; nt < 4; ++nt)
#pragma unroll
        for (int r = 0; r < 8; ++r) acc[nt][r] = 0.0f;

    const unsigned int* arowp = la[wm * 16 + l16];

    for (int k0 = 0; k0 < K; k0 += 64) {
        const int kc = k0 >> 1;            // dword offset into packed row

        // stage A tile (32x32 dw) + W tile (256x32 dw): async global->LDS b128
        async_copy_b128(laAddr, &A32[(size_t)(m0 + arow) * K2 + kc + acc4]);
#pragma unroll
        for (int j = 0; j < 8; ++j)
            async_copy_b128(lbAddr[j], &W32[(size_t)(n0 + lbRow[j]) * K2 + kc + lbCol[j]]);
        wait_async0();
        __syncthreads();

#pragma unroll
        for (int ks = 0; ks < 2; ++ks) {
            const int kb = ks * 16;        // dword base of this 32-K sub-tile
            // A fragment: 16x32 bf16 (two aligned 4-dword runs)
            v16bf av = frag_from_lds(arowp, kb + half * 8, kb + half * 8 + 4);
            // B fragments: 32x16 bf16 each (8 consecutive dwords)
            v16bf bv0 = frag_from_lds(lb[wn * 64 +  0 + l16], kb + half * 8, kb + half * 8 + 4);
            v16bf bv1 = frag_from_lds(lb[wn * 64 + 16 + l16], kb + half * 8, kb + half * 8 + 4);
            v16bf bv2 = frag_from_lds(lb[wn * 64 + 32 + l16], kb + half * 8, kb + half * 8 + 4);
            v16bf bv3 = frag_from_lds(lb[wn * 64 + 48 + l16], kb + half * 8, kb + half * 8 + 4);
            acc[0] = __builtin_amdgcn_wmma_f32_16x16x32_bf16(false, av, false, bv0, (short)0, acc[0], false, false);
            acc[1] = __builtin_amdgcn_wmma_f32_16x16x32_bf16(false, av, false, bv1, (short)0, acc[1], false, false);
            acc[2] = __builtin_amdgcn_wmma_f32_16x16x32_bf16(false, av, false, bv2, (short)0, acc[2], false, false);
            acc[3] = __builtin_amdgcn_wmma_f32_16x16x32_bf16(false, av, false, bv3, (short)0, acc[3], false, false);
        }
        __syncthreads();
    }

    const int halfC = N >> 1;
#pragma unroll
    for (int nt = 0; nt < 4; ++nt) {
#pragma unroll
        for (int r = 0; r < 8; ++r) {
            int row = m0 + wm * 16 + r + half * 8;
            int col = n0 + wn * 64 + nt * 16 + l16;
            float val = acc[nt][r] + bias[col];
            if (epi == 1)      val = val / (1.0f + __expf(-val));                       // silu
            else if (epi == 2) val = (val > 0.0f) ? (val + 1.0f) : __expf(val);         // elu+1
            else if (epi == 3) val = 0.5f * val * (1.0f + erff(val * 0.70710678f));     // gelu
            if (epi == 4) {
                if (col < halfC) outF [(size_t)row * halfC + col] =
                                     (val > 0.0f) ? (val + 1.0f) : __expf(val);         // k
                else             outF2[(size_t)row * halfC + (col - halfC)] = val;      // v
            } else {
                size_t o = (size_t)row * N + col;
                if (resid) val += resid[o];
                if (outF)  outF[o] = val;
                if (outB)  outB[o] = f32_to_bf16(val);
            }
        }
    }
}

// ---- weight prep: fp32 [K,N] -> bf16 [N,K] (transpose + convert) -----------
__global__ __launch_bounds__(256)
void k_wprep(const float* __restrict__ in, unsigned short* __restrict__ out,
             int K, int N)
{
    __shared__ float s[32][33];
    int n0 = blockIdx.x * 32, k0 = blockIdx.y * 32;
    int t = threadIdx.x;
#pragma unroll
    for (int j = 0; j < 4; ++j) {
        int idx = t + j * 256;
        int r = idx >> 5, c = idx & 31;
        s[r][c] = in[(size_t)(k0 + r) * N + n0 + c];
    }
    __syncthreads();
#pragma unroll
    for (int j = 0; j < 4; ++j) {
        int idx = t + j * 256;
        int r = idx >> 5, c = idx & 31;        // r: n offset, c: k offset
        out[(size_t)(n0 + r) * K + k0 + c] = f32_to_bf16(s[c][r]);
    }
}

__global__ void k_cast_bf16(const float* __restrict__ in,
                            unsigned short* __restrict__ out)
{
    size_t i = (size_t)blockIdx.x * blockDim.x + threadIdx.x;
    out[i] = f32_to_bf16(in[i]);
}

__global__ void k_mul_bf16(const float* __restrict__ a, const float* __restrict__ b,
                           unsigned short* __restrict__ out)
{
    size_t i = (size_t)blockIdx.x * blockDim.x + threadIdx.x;
    out[i] = f32_to_bf16(a[i] * b[i]);
}

// ---- depthwise conv3 along token dim: out = add + conv3(src) + bias --------
__global__ void k_conv3(const float* __restrict__ src, const float* __restrict__ w,
                        const float* __restrict__ bias, const float* __restrict__ add,
                        float* __restrict__ out, int L, int C)
{
    size_t idx = (size_t)blockIdx.x * blockDim.x + threadIdx.x;
    int c = (int)(idx % C);
    int l = (int)((idx / C) % L);
    float xm = (l > 0)     ? src[idx - C] : 0.0f;
    float x0 = src[idx];
    float xp = (l < L - 1) ? src[idx + C] : 0.0f;
    float y = w[c * 3] * xm + w[c * 3 + 1] * x0 + w[c * 3 + 2] * xp + bias[c];
    if (add) y += add[idx];
    out[idx] = y;
}

// ---- dwc branch: flat (b,l,c)->(b,c,l) reinterpret, conv3 along l, silu ----
__global__ void k_dwc_silu(const float* __restrict__ src, const float* __restrict__ w,
                           const float* __restrict__ bias,
                           unsigned short* __restrict__ out, int L, int C)
{
    size_t idx = (size_t)blockIdx.x * blockDim.x + threadIdx.x;
    size_t LC = (size_t)L * C;
    size_t p = idx % LC;
    int ch  = (int)(p / L);
    int pos = (int)(p % L);
    float xm = (pos > 0)     ? src[idx - 1] : 0.0f;
    float x0 = src[idx];
    float xp = (pos < L - 1) ? src[idx + 1] : 0.0f;
    float y = w[ch * 3] * xm + w[ch * 3 + 1] * x0 + w[ch * 3 + 2] * xp + bias[ch];
    y = y / (1.0f + __expf(-y));
    out[idx] = f32_to_bf16(y);
}

// ---- LayerNorm over C=512, one block per token, bf16 out -------------------
__global__ __launch_bounds__(256)
void k_ln(const float* __restrict__ x, const float* __restrict__ g,
          const float* __restrict__ b, unsigned short* __restrict__ out, int C)
{
    __shared__ float s1[256], s2[256];
    int t = threadIdx.x;
    size_t base = (size_t)blockIdx.x * C;
    float v0 = x[base + t], v1 = x[base + t + 256];
    s1[t] = v0 + v1;
    s2[t] = v0 * v0 + v1 * v1;
    __syncthreads();
    for (int s = 128; s > 0; s >>= 1) {
        if (t < s) { s1[t] += s1[t + s]; s2[t] += s2[t + s]; }
        __syncthreads();
    }
    float mu = s1[0] / (float)C;
    float var = s2[0] / (float)C - mu * mu;
    float rs = rsqrtf(var + 1e-5f);
    out[base + t]       = f32_to_bf16((v0 - mu) * rs * g[t]       + b[t]);
    out[base + t + 256] = f32_to_bf16((v1 - mu) * rs * g[t + 256] + b[t + 256]);
}

// ---- k_mean over tokens, per (b,h,d) ---------------------------------------
__global__ __launch_bounds__(256)
void k_kmean(const float* __restrict__ k, float* __restrict__ km, int L, int C, int H)
{
    int bh = blockIdx.x, b = bh / H, h = bh % H;
    int t = threadIdx.x, d = t & 63, seg = t >> 6;
    const float* base = k + (size_t)b * L * C + h * 64 + d;
    int npart = L / 4;
    float s = 0.0f;
    for (int n = seg * npart; n < (seg + 1) * npart; ++n) s += base[(size_t)n * C];
    __shared__ float red[256];
    red[t] = s;
    __syncthreads();
    if (t < 64) {
        float tot = red[t] + red[t + 64] + red[t + 128] + red[t + 192];
        km[(size_t)bh * 64 + t] = tot / (float)L;
    }
}

// ---- kv_mat[b,h] = (1/L) * K^T V  (64x64 per head) -------------------------
__global__ __launch_bounds__(256)
void k_kvmat(const float* __restrict__ kk, const float* __restrict__ vv,
             float* __restrict__ kvm, int L, int C, int H)
{
    int bh = blockIdx.x, b = bh / H, h = bh % H;
    int t = threadIdx.x;
    __shared__ float kt[32][64];
    __shared__ float vt[32][68];
    int d  = t >> 2;
    int eb = (t & 3) * 16;
    float acc[16];
#pragma unroll
    for (int e = 0; e < 16; ++e) acc[e] = 0.0f;
    const float* kbase = kk + (size_t)b * L * C + h * 64;
    const float* vbase = vv + (size_t)b * L * C + h * 64;
    for (int n0 = 0; n0 < L; n0 += 32) {
#pragma unroll
        for (int j = 0; j < 8; ++j) {
            int idx = t + j * 256;
            int i = idx >> 6, dd = idx & 63;
            kt[i][dd] = kbase[(size_t)(n0 + i) * C + dd];
            vt[i][dd] = vbase[(size_t)(n0 + i) * C + dd];
        }
        __syncthreads();
#pragma unroll 8
        for (int i = 0; i < 32; ++i) {
            float kd = kt[i][d];
#pragma unroll
            for (int e = 0; e < 16; ++e) acc[e] += kd * vt[i][eb + e];
        }
        __syncthreads();
    }
    float sc = 1.0f / (float)L;
#pragma unroll
    for (int e = 0; e < 16; ++e)
        kvm[((size_t)bh * 64 + d) * 64 + eb + e] = acc[e] * sc;
}

// ---- out[n,e] = z[n] * q[n,:] @ kv_mat, 64 tokens per block ----------------
__global__ __launch_bounds__(256)
void k_attn_out(const float* __restrict__ q, const float* __restrict__ kvm,
                const float* __restrict__ km, float* __restrict__ xa,
                int L, int C, int H)
{
    const int nb = L / 64;
    int blk = blockIdx.x;
    int bh = blk / nb, n0 = (blk % nb) * 64;
    int b = bh / H, h = bh % H;
    int t = threadIdx.x;
    __shared__ float kvs[64][65];
    __shared__ float qs[64][65];
    __shared__ float kms[64];
    __shared__ float zs[64];
    const float* qbase = q + (size_t)b * L * C + h * 64;
#pragma unroll
    for (int j = 0; j < 16; ++j) {
        int idx = t + j * 256;
        int i = idx >> 6, dd = idx & 63;
        kvs[i][dd] = kvm[(size_t)bh * 4096 + idx];
        qs[i][dd]  = qbase[(size_t)(n0 + i) * C + dd];
    }
    if (t < 64) kms[t] = km[(size_t)bh * 64 + t];
    __syncthreads();
    if (t < 64) {
        float s = 0.0f;
#pragma unroll 16
        for (int dd = 0; dd < 64; ++dd) s += qs[t][dd] * kms[dd];
        zs[t] = 1.0f / (s + 1e-6f);
    }
    __syncthreads();
    int i = t >> 2, eb = (t & 3) * 16;
    float acc[16];
#pragma unroll
    for (int e = 0; e < 16; ++e) acc[e] = 0.0f;
    for (int dd = 0; dd < 64; ++dd) {
        float qv = qs[i][dd];
#pragma unroll
        for (int e = 0; e < 16; ++e) acc[e] += qv * kvs[dd][eb + e];
    }
    float z = zs[i];
    float* obase = xa + (size_t)b * L * C + (size_t)(n0 + i) * C + h * 64 + eb;
#pragma unroll
    for (int e = 0; e < 16; ++e) obase[e] = acc[e] * z;
}

// ---------------------------------------------------------------------------
extern "C" void kernel_launch(void* const* d_in, const int* in_sizes, int n_in,
                              void* d_out, int out_size, void* d_ws, size_t ws_size,
                              hipStream_t stream)
{
    const int Bb = 8, Ll = 2048, Cc = 512, Hh = 8;
    const size_t NTOK = (size_t)Bb * Ll;          // 16384
    const size_t NC = NTOK * Cc;                  // 8388608

    const float* x_q    = (const float*)d_in[0];
    const float* x_kv   = (const float*)d_in[1];
    const float* cpe1_w = (const float*)d_in[2];
    const float* cpe1_b = (const float*)d_in[3];
    const float* norm1_g= (const float*)d_in[4];
    const float* norm1_b= (const float*)d_in[5];
    const float* in_w   = (const float*)d_in[6];
    const float* in_b   = (const float*)d_in[7];
    const float* act_w  = (const float*)d_in[8];
    const float* act_b  = (const float*)d_in[9];
    const float* dwc_w  = (const float*)d_in[10];
    const float* dwc_b  = (const float*)d_in[11];
    const float* q_w    = (const float*)d_in[12];
    const float* q_b    = (const float*)d_in[13];
    const float* kv_w   = (const float*)d_in[14];
    const float* kv_b   = (const float*)d_in[15];
    const float* lepe_w = (const float*)d_in[16];
    const float* lepe_b = (const float*)d_in[17];
    const float* out_w  = (const float*)d_in[18];
    const float* out_b  = (const float*)d_in[19];
    const float* cpe2_w = (const float*)d_in[20];
    const float* cpe2_b = (const float*)d_in[21];
    const float* norm2_g= (const float*)d_in[22];
    const float* norm2_b= (const float*)d_in[23];
    const float* fc1_w  = (const float*)d_in[24];
    const float* fc1_b  = (const float*)d_in[25];
    const float* fc2_w  = (const float*)d_in[26];
    const float* fc2_b  = (const float*)d_in[27];

    // -------- workspace layout --------
    char* base = (char*)d_ws;
    size_t off = 0;
    auto alloc = [&](size_t bytes) { size_t o = off; off += (bytes + 255) & ~(size_t)255; return o; };
    float* bufA = (float*)(base + alloc(NC * 4));   // shortcut / x3
    float* bufC = (float*)(base + alloc(NC * 4));   // act_res / x2
    float* bufD = (float*)(base + alloc(NC * 4));   // xp_pre  / xa
    float* bufQ = (float*)(base + alloc(NC * 4));   // q   (later: hbuf low half)
    float* bufK = (float*)(base + alloc(NC * 4));   // k   (later: hbuf high half)
    float* bufV = (float*)(base + alloc(NC * 4));   // v
    unsigned short* bbuf = (unsigned short*)(base + alloc(NC * 2));  // rotating bf16
    unsigned short* xqb  = (unsigned short*)(base + alloc(NC * 2));  // bf16(x_q)
    unsigned short* wt_act = (unsigned short*)(base + alloc((size_t)512 * 512 * 2));
    unsigned short* wt_in  = (unsigned short*)(base + alloc((size_t)512 * 512 * 2));
    unsigned short* wt_q   = (unsigned short*)(base + alloc((size_t)512 * 512 * 2));
    unsigned short* wt_kv  = (unsigned short*)(base + alloc((size_t)512 * 1024 * 2));
    unsigned short* wt_out = (unsigned short*)(base + alloc((size_t)512 * 512 * 2));
    unsigned short* wt_fc1 = (unsigned short*)(base + alloc((size_t)512 * 2048 * 2));
    unsigned short* wt_fc2 = (unsigned short*)(base + alloc((size_t)2048 * 512 * 2));
    float* kmean = (float*)(base + alloc((size_t)Bb * Hh * 64 * 4));
    float* kvm   = (float*)(base + alloc((size_t)Bb * Hh * 4096 * 4));
    unsigned short* hbuf = (unsigned short*)bufQ;   // 16384x2048 bf16 == bufQ+bufK

    const int blocksNC = (int)(NC / 256);           // 32768

    // -------- weight prep (fp32 [K,N] -> bf16 [N,K]) --------
    k_wprep<<<dim3(512 / 32, 512 / 32), 256, 0, stream>>>(act_w, wt_act, 512, 512);
    k_wprep<<<dim3(512 / 32, 512 / 32), 256, 0, stream>>>(in_w,  wt_in,  512, 512);
    k_wprep<<<dim3(512 / 32, 512 / 32), 256, 0, stream>>>(q_w,   wt_q,   512, 512);
    k_wprep<<<dim3(1024 / 32, 512 / 32), 256, 0, stream>>>(kv_w, wt_kv,  512, 1024);
    k_wprep<<<dim3(512 / 32, 512 / 32), 256, 0, stream>>>(out_w, wt_out, 512, 512);
    k_wprep<<<dim3(2048 / 32, 512 / 32), 256, 0, stream>>>(fc1_w, wt_fc1, 512, 2048);
    k_wprep<<<dim3(512 / 32, 2048 / 32), 256, 0, stream>>>(fc2_w, wt_fc2, 2048, 512);
    k_cast_bf16<<<blocksNC, 256, 0, stream>>>(x_q, xqb);

    // x = x_kv + cpe1(x_kv)
    k_conv3<<<blocksNC, 256, 0, stream>>>(x_kv, cpe1_w, cpe1_b, x_kv, bufA, Ll, Cc);
    // x_ln = LN(x) -> bf16
    k_ln<<<(int)NTOK, 256, 0, stream>>>(bufA, norm1_g, norm1_b, bbuf, Cc);
    // act_res = silu(x_ln @ act_w + b)
    k_gemm_bf16<<<dim3(NTOK / 32, 512 / 256), 256, 0, stream>>>(
        (const unsigned int*)bbuf, (const unsigned int*)wt_act, act_b,
        bufC, nullptr, nullptr, nullptr, (int)NTOK, 512, 512, 1);
    // xp_pre = x_ln @ in_w + b
    k_gemm_bf16<<<dim3(NTOK / 32, 512 / 256), 256, 0, stream>>>(
        (const unsigned int*)bbuf, (const unsigned int*)wt_in, in_b,
        bufD, nullptr, nullptr, nullptr, (int)NTOK, 512, 512, 0);
    // xp = silu(dwc(reinterpret(xp_pre))) -> bf16
    k_dwc_silu<<<blocksNC, 256, 0, stream>>>(bufD, dwc_w, dwc_b, bbuf, Ll, Cc);
    // q = elu(x_q @ q_w + b) + 1
    k_gemm_bf16<<<dim3(NTOK / 32, 512 / 256), 256, 0, stream>>>(
        (const unsigned int*)xqb, (const unsigned int*)wt_q, q_b,
        bufQ, nullptr, nullptr, nullptr, (int)NTOK, 512, 512, 2);
    // kv = xp @ kv_w + b ; k = elu(kv[:,:C])+1 ; v = kv[:,C:]
    k_gemm_bf16<<<dim3(NTOK / 32, 1024 / 256), 256, 0, stream>>>(
        (const unsigned int*)bbuf, (const unsigned int*)wt_kv, kv_b,
        bufK, bufV, nullptr, nullptr, (int)NTOK, 1024, 512, 4);
    // attention core
    k_kmean<<<Bb * Hh, 256, 0, stream>>>(bufK, kmean, Ll, Cc, Hh);
    k_kvmat<<<Bb * Hh, 256, 0, stream>>>(bufK, bufV, kvm, Ll, Cc, Hh);
    k_attn_out<<<Bb * Hh * (Ll / 64), 256, 0, stream>>>(bufQ, kvm, kmean, bufD, Ll, Cc, Hh);
    // xa += lepe(v)
    k_conv3<<<blocksNC, 256, 0, stream>>>(bufV, lepe_w, lepe_b, bufD, bufD, Ll, Cc);
    // (xa * act_res) -> bf16
    k_mul_bf16<<<blocksNC, 256, 0, stream>>>(bufD, bufC, bbuf);
    // x2 = shortcut + (xa*act_res) @ out_w + b
    k_gemm_bf16<<<dim3(NTOK / 32, 512 / 256), 256, 0, stream>>>(
        (const unsigned int*)bbuf, (const unsigned int*)wt_out, out_b,
        bufC, nullptr, nullptr, bufA, (int)NTOK, 512, 512, 0);
    // x3 = x2 + cpe2(x2)
    k_conv3<<<blocksNC, 256, 0, stream>>>(bufC, cpe2_w, cpe2_b, bufC, bufA, Ll, Cc);
    // h = LN(x3) -> bf16
    k_ln<<<(int)NTOK, 256, 0, stream>>>(bufA, norm2_g, norm2_b, bbuf, Cc);
    // h1 = gelu(h @ fc1_w + b) -> bf16
    k_gemm_bf16<<<dim3(NTOK / 32, 2048 / 256), 256, 0, stream>>>(
        (const unsigned int*)bbuf, (const unsigned int*)wt_fc1, fc1_b,
        nullptr, nullptr, hbuf, nullptr, (int)NTOK, 2048, 512, 3);
    // out = x3 + h1 @ fc2_w + b
    k_gemm_bf16<<<dim3(NTOK / 32, 512 / 256), 256, 0, stream>>>(
        (const unsigned int*)hbuf, (const unsigned int*)wt_fc2, fc2_b,
        (float*)d_out, nullptr, nullptr, bufA, (int)NTOK, 512, 2048, 0);
}
